// DiscreteBundleSheafDiffusion_49340584296531
// MI455X (gfx1250) — matrix operand order repaired and checked by reading, not compiled, not run on hardware
//
#include <hip/hip_runtime.h>
#include <hip/hip_bf16.h>

// ---------------------------------------------------------------------------
// DiscreteBundleSheafDiffusion forward, CDNA5 (gfx1250, wave32, WMMA)
//
// Sizes (fixed by reference): N=50000, NE=400000 directed edges, IN=128,
// FD=3, HID=32 (FD*HID=96), OUT=32, 2 layers.
// ---------------------------------------------------------------------------

typedef __attribute__((ext_vector_type(16))) _Float16 v16h;
typedef __attribute__((ext_vector_type(8)))  _Float16 v8h;
typedef __attribute__((ext_vector_type(8)))  float    v8f;

#define FD   3
#define HID  32
#define FH   96      // FD*HID
#define IN_CH 128
#define OUT_CH 32

static inline int cdiv(long a, long b) { return (int)((a + b - 1) / b); }

// ---------------------------------------------------------------------------
// generic helpers
// ---------------------------------------------------------------------------
__global__ void fill_f32(float* __restrict__ p, int n, float v) {
    int i = blockIdx.x * blockDim.x + threadIdx.x;
    if (i < n) p[i] = v;
}

__global__ void cvt_f32_to_f16(const float* __restrict__ in,
                               _Float16* __restrict__ out, int n) {
    int i = blockIdx.x * blockDim.x + threadIdx.x;
    if (i < n) out[i] = (_Float16)in[i];
}

// Pre-swizzle a row-major weight W[Nout, Kdim] into WMMA B-operand register
// layout (B = W^T, K x N). Tile (kt,nt): lane L holds column n = nt*16+(L&15),
// halves h=0..15 hold k = kt*32 + (L<16 ? 0 : 16) + h.
__global__ void swizzle_b_f16(const float* __restrict__ W,
                              _Float16* __restrict__ Bsw,
                              int Kdim, int KT, int NT) {
    int idx = blockIdx.x * blockDim.x + threadIdx.x;
    int total = KT * NT * 512;
    if (idx >= total) return;
    int h    = idx & 15;
    int lane = (idx >> 4) & 31;
    int t    = idx >> 9;            // kt*NT + nt
    int nt   = t % NT;
    int kt   = t / NT;
    int k = kt * 32 + ((lane & 16) ? 16 : 0) + h;
    int n = nt * 16 + (lane & 15);
    Bsw[idx] = (_Float16)W[(size_t)n * Kdim + k];
}

// ---------------------------------------------------------------------------
// WMMA GEMM: Out[M, NT*16] = A[M, KT*32] * Bsw^ + epilogue
// One wave per 16-row M tile; A kept in registers across all N tiles.
// EPI: 0 = store, 1 = +bias, ELU, 2 = +bias
// ---------------------------------------------------------------------------
template <int KT, int NT, int EPI>
__global__ void __launch_bounds__(32)
gemm_wmma_f16(const _Float16* __restrict__ A,
              const _Float16* __restrict__ Bsw,
              const float* __restrict__ bias,
              float* __restrict__ Out) {
    const int lane  = threadIdx.x;
    const int mt    = blockIdx.x;
    const int Kdim  = KT * 32;
    const int Ncols = NT * 16;
    const int mrow  = lane & 15;
    const int kbA   = (lane < 16) ? 0 : 16;  // A: K base offset per half-wave

    v8f acc[NT] = {};

    const _Float16* arow = A + (size_t)(mt * 16 + mrow) * Kdim;

#pragma unroll
    for (int kt = 0; kt < KT; ++kt) {
        // A 16x32 f16 tile: two contiguous 128-bit loads per lane
        v8h x0 = *(const v8h*)(arow + kt * 32 + kbA);
        v8h x1 = *(const v8h*)(arow + kt * 32 + kbA + 8);
        v16h av;
#pragma unroll
        for (int i = 0; i < 8; ++i) { av[i] = x0[i]; av[8 + i] = x1[i]; }
#pragma unroll
        for (int nt = 0; nt < NT; ++nt) {
            const v16h bv =
                *(const v16h*)(Bsw + ((((kt * NT + nt) * 32) + lane) << 4));
            acc[nt] = __builtin_amdgcn_wmma_f32_16x16x32_f16(
                false, av, false, bv, (short)0, acc[nt], false, false);
        }
    }

    const int rbase = (lane < 16) ? 0 : 8;   // C/D layout
#pragma unroll
    for (int nt = 0; nt < NT; ++nt) {
#pragma unroll
        for (int r = 0; r < 8; ++r) {
            int m = mt * 16 + rbase + r;
            int n = nt * 16 + mrow;
            float v = acc[nt][r];
            if constexpr (EPI >= 1) v += bias[n];
            if constexpr (EPI == 1) v = (v > 0.f) ? v : expm1f(v);
            Out[(size_t)m * Ncols + n] = v;
        }
    }
}

// ---------------------------------------------------------------------------
// Spectral normalization (power iteration, detached u/v) for Wl(3x3), Wr(32x32)
// One wave per layer. Writes WlN[l][9] and WrN[l][1024].
// ---------------------------------------------------------------------------
__global__ void __launch_bounds__(32)
specnorm_kernel(const float* __restrict__ Wl_all,
                const float* __restrict__ Wr_all,
                float* __restrict__ WlN, float* __restrict__ WrN) {
    __shared__ float Ws[1024];
    const int l = blockIdx.x;
    const int lane = threadIdx.x;
    const float* Wr = Wr_all + l * 1024;
    for (int i = lane; i < 1024; i += 32) Ws[i] = Wr[i];
    __syncthreads();

    float u = 0.17677669529f;   // 1/sqrt(32)
    float v = 0.f;
    for (int it = 0; it < 20; ++it) {
        float a = 0.f;
        for (int i = 0; i < 32; ++i) a += Ws[i * 32 + lane] * __shfl(u, i);
        float n2 = a * a;
        for (int o = 16; o; o >>= 1) n2 += __shfl_xor(n2, o);
        v = a / (sqrtf(n2) + 1e-12f);
        a = 0.f;
        for (int j = 0; j < 32; ++j) a += Ws[lane * 32 + j] * __shfl(v, j);
        n2 = a * a;
        for (int o = 16; o; o >>= 1) n2 += __shfl_xor(n2, o);
        u = a / (sqrtf(n2) + 1e-12f);
    }
    {   // final v, sigma = u^T W v
        float a = 0.f;
        for (int i = 0; i < 32; ++i) a += Ws[i * 32 + lane] * __shfl(u, i);
        float n2 = a * a;
        for (int o = 16; o; o >>= 1) n2 += __shfl_xor(n2, o);
        v = a / (sqrtf(n2) + 1e-12f);
        a = 0.f;
        for (int j = 0; j < 32; ++j) a += Ws[lane * 32 + j] * __shfl(v, j);
        float sig = u * a;
        for (int o = 16; o; o >>= 1) sig += __shfl_xor(sig, o);
        float inv = 1.f / sig;
        for (int i = lane; i < 1024; i += 32) WrN[l * 1024 + i] = Ws[i] * inv;
    }

    if (lane == 0) {   // 3x3 Wl, scalar
        float W[9];
        for (int i = 0; i < 9; ++i) W[i] = Wl_all[l * 9 + i];
        float uu[3] = {0.57735027f, 0.57735027f, 0.57735027f};
        float vv[3];
        for (int it = 0; it < 20; ++it) {
            float n2 = 0.f;
            for (int j = 0; j < 3; ++j) {
                vv[j] = W[j] * uu[0] + W[3 + j] * uu[1] + W[6 + j] * uu[2];
                n2 += vv[j] * vv[j];
            }
            float inv = 1.f / (sqrtf(n2) + 1e-12f);
            for (int j = 0; j < 3; ++j) vv[j] *= inv;
            n2 = 0.f;
            for (int i = 0; i < 3; ++i) {
                uu[i] = W[i * 3] * vv[0] + W[i * 3 + 1] * vv[1] + W[i * 3 + 2] * vv[2];
                n2 += uu[i] * uu[i];
            }
            inv = 1.f / (sqrtf(n2) + 1e-12f);
            for (int i = 0; i < 3; ++i) uu[i] *= inv;
        }
        float n2 = 0.f;
        for (int j = 0; j < 3; ++j) {
            vv[j] = W[j] * uu[0] + W[3 + j] * uu[1] + W[6 + j] * uu[2];
            n2 += vv[j] * vv[j];
        }
        float inv = 1.f / (sqrtf(n2) + 1e-12f);
        for (int j = 0; j < 3; ++j) vv[j] *= inv;
        float sig = 0.f;
        for (int i = 0; i < 3; ++i) {
            float t = W[i * 3] * vv[0] + W[i * 3 + 1] * vv[1] + W[i * 3 + 2] * vv[2];
            sig += uu[i] * t;
        }
        inv = 1.f / sig;
        for (int i = 0; i < 9; ++i) WlN[l * 9 + i] = W[i] * inv;
    }
}

// ---------------------------------------------------------------------------
// Per-node dot products against sheaf row-1 and edge-weight row:
// nd[n] = { xn.ws[:96], xn.ws[96:], xn.wt[:96], xn.wt[96:] }
// One wave per node.
// ---------------------------------------------------------------------------
__global__ void __launch_bounds__(128)
node_dots(const float* __restrict__ xc,
          const float* __restrict__ wsheaf_row,   // 192
          const float* __restrict__ wwt_row,      // 192
          float4* __restrict__ nd, int n_nodes) {
    int wave = threadIdx.x >> 5;
    int lane = threadIdx.x & 31;
    int n = blockIdx.x * 4 + wave;
    if (n >= n_nodes) return;
    const float* xr = xc + (size_t)n * FH;
    float sa = 0.f, sb = 0.f, wa = 0.f, wb = 0.f;
#pragma unroll
    for (int i = lane; i < FH; i += 32) {
        float xv = xr[i];
        sa += xv * wsheaf_row[i];
        sb += xv * wsheaf_row[FH + i];
        wa += xv * wwt_row[i];
        wb += xv * wwt_row[FH + i];
    }
    for (int o = 16; o; o >>= 1) {
        sa += __shfl_xor(sa, o); sb += __shfl_xor(sb, o);
        wa += __shfl_xor(wa, o); wb += __shfl_xor(wb, o);
    }
    if (lane == 0) nd[n] = make_float4(sa, sb, wa, wb);
}

// ---------------------------------------------------------------------------
// Per-edge: t, t_rev via Cayley (only p[:,1] survives A = M - M^T for D=2),
// symmetric weight w = sigmoid(fwd)*sigmoid(rev), transport T = Q_e^T Q_rev
// (rotation combine). Stores {Tc, Ts, w^2} per edge, accumulates deg.
// ---------------------------------------------------------------------------
__global__ void edge_prep(const int* __restrict__ ei,
                          const float4* __restrict__ nd,
                          float4* __restrict__ edg,
                          float* __restrict__ deg, int ne) {
    int e = blockIdx.x * blockDim.x + threadIdx.x;
    if (e >= ne) return;
    int r = ei[e];
    int c = ei[ne + e];
    float4 fr = nd[r];
    float4 fc = nd[c];
    float t  = tanhf(fr.x + fc.y);
    float tr = tanhf(fc.x + fr.y);
    float wd = 1.f / (1.f + expf(-(fr.z + fc.w)));
    float wr = 1.f / (1.f + expf(-(fc.z + fr.w)));
    float w  = wd * wr;
    float w2 = w * w;
    float i1 = 1.f / (1.f + t * t);
    float ce = (1.f - t * t) * i1, se = 2.f * t * i1;
    float i2 = 1.f / (1.f + tr * tr);
    float cr = (1.f - tr * tr) * i2, sr = 2.f * tr * i2;
    float Tc = ce * cr + se * sr;           // (Q_e^T Q_rev) rotation
    float Ts = ce * sr - se * cr;
    edg[e] = make_float4(Tc, Ts, w2, 0.f);
    atomicAdd(&deg[r], w2);
}

__global__ void dinv_kernel(const float* __restrict__ deg,
                            float* __restrict__ dinv, int n) {
    int i = blockIdx.x * blockDim.x + threadIdx.x;
    if (i >= n) return;
    float d = deg[i];
    dinv[i] = (d > 0.f) ? rsqrtf(fmaxf(d, 1e-30f)) : 0.f;
}

// ---------------------------------------------------------------------------
// Left (stalk) mix with spectral-normalized 3x3, output f16 (WMMA A operand).
// One thread per (node, channel).
// ---------------------------------------------------------------------------
__global__ void wl_mix(const float* __restrict__ xc,
                       const float* __restrict__ wln,   // 9
                       _Float16* __restrict__ out, int n_nodes) {
    int idx = blockIdx.x * blockDim.x + threadIdx.x;
    if (idx >= n_nodes * HID) return;
    int n = idx >> 5;
    int h = idx & 31;
    const float* xr = xc + (size_t)n * FH + h;
    float a0 = xr[0], a1 = xr[HID], a2 = xr[2 * HID];
    _Float16* orow = out + (size_t)n * FH + h;
    orow[0]       = (_Float16)(wln[0] * a0 + wln[1] * a1 + wln[2] * a2);
    orow[HID]     = (_Float16)(wln[3] * a0 + wln[4] * a1 + wln[5] * a2);
    orow[2 * HID] = (_Float16)(wln[6] * a0 + wln[7] * a1 + wln[8] * a2);
}

// ---------------------------------------------------------------------------
// Laplacian scatter: agg[row] += cnorm * [T @ y[col][:2]; y[col][2]]
// 8 threads per edge (4 channels each), 12 f32 atomics per thread.
// ---------------------------------------------------------------------------
__global__ void aggregate(const int* __restrict__ ei,
                          const float4* __restrict__ edg,
                          const float* __restrict__ dinv,
                          const float* __restrict__ y,
                          float* __restrict__ agg, int ne) {
    int idx = blockIdx.x * blockDim.x + threadIdx.x;
    if (idx >= ne * 8) return;
    int e = idx >> 3;
    int h = (idx & 7) * 4;
    int r = ei[e];
    int c = ei[ne + e];
    float4 t4 = edg[e];
    float cn = t4.z * dinv[r] * dinv[c];
    if (cn == 0.f) return;
    const float* yc = y + (size_t)c * FH + h;
    float4 y0 = *(const float4*)(yc);
    float4 y1 = *(const float4*)(yc + HID);
    float4 y2 = *(const float4*)(yc + 2 * HID);
    float Tc = t4.x, Ts = t4.y;
    float* ar = agg + (size_t)r * FH + h;
    atomicAdd(ar + 0,  cn * (Tc * y0.x - Ts * y1.x));
    atomicAdd(ar + 1,  cn * (Tc * y0.y - Ts * y1.y));
    atomicAdd(ar + 2,  cn * (Tc * y0.z - Ts * y1.z));
    atomicAdd(ar + 3,  cn * (Tc * y0.w - Ts * y1.w));
    atomicAdd(ar + HID + 0, cn * (Ts * y0.x + Tc * y1.x));
    atomicAdd(ar + HID + 1, cn * (Ts * y0.y + Tc * y1.y));
    atomicAdd(ar + HID + 2, cn * (Ts * y0.z + Tc * y1.z));
    atomicAdd(ar + HID + 3, cn * (Ts * y0.w + Tc * y1.w));
    atomicAdd(ar + 2 * HID + 0, cn * y2.x);
    atomicAdd(ar + 2 * HID + 1, cn * y2.y);
    atomicAdd(ar + 2 * HID + 2, cn * y2.z);
    atomicAdd(ar + 2 * HID + 3, cn * y2.w);
}

// ---------------------------------------------------------------------------
// z = ELU(diag*y - agg); xc = (1+tanh(eps_f)) * xc - z
// ---------------------------------------------------------------------------
__global__ void combine(const float* __restrict__ deg,
                        const float* __restrict__ y,
                        const float* __restrict__ agg,
                        const float* __restrict__ eps,   // layer base, 3 vals
                        float* __restrict__ xc, int n_nodes) {
    int idx = blockIdx.x * blockDim.x + threadIdx.x;
    if (idx >= n_nodes * FH) return;
    int n = idx / FH;
    int f = (idx - n * FH) >> 5;
    float diag = (deg[n] > 0.f) ? 1.f : 0.f;
    float v = diag * y[idx] - agg[idx];
    float z = (v > 0.f) ? v : expm1f(v);
    float coeff = 1.f + tanhf(eps[f]);
    xc[idx] = coeff * xc[idx] - z;
}

// ---------------------------------------------------------------------------
// Host-side orchestration
// ---------------------------------------------------------------------------
extern "C" void kernel_launch(void* const* d_in, const int* in_sizes, int n_in,
                              void* d_out, int out_size, void* d_ws, size_t ws_size,
                              hipStream_t stream) {
    const float* x   = (const float*)d_in[0];
    const int*   ei  = (const int*)d_in[1];
    const float* W1  = (const float*)d_in[2];
    const float* b1  = (const float*)d_in[3];
    const float* W2  = (const float*)d_in[4];
    const float* b2  = (const float*)d_in[5];
    const float* Wl  = (const float*)d_in[6];   // [2,3,3]
    const float* Wr  = (const float*)d_in[7];   // [2,32,32]
    const float* eps = (const float*)d_in[8];   // [2,3,1]
    const float* Wsh = (const float*)d_in[9];   // [2,3,192]
    const float* Wwt = (const float*)d_in[10];  // [2,1,192]
    float* out = (float*)d_out;

    const int N  = in_sizes[0] / IN_CH;   // 50000
    const int NE = in_sizes[1] / 2;       // 400000 directed edges

    // workspace carve-up (256B aligned)
    char* p = (char*)d_ws;
    auto carve = [&](size_t bytes) -> void* {
        void* r = (void*)p;
        p += (bytes + 255) & ~(size_t)255;
        return r;
    };
    float*    XC   = (float*)carve((size_t)N * FH * 4);
    float*    Y    = (float*)carve((size_t)N * FH * 4);
    float*    AGG  = (float*)carve((size_t)N * FH * 4);
    _Float16* MXH  = (_Float16*)carve((size_t)N * FH * 2);
    _Float16* XH   = (_Float16*)carve((size_t)N * IN_CH * 2);
    float4*   EDG  = (float4*)carve((size_t)NE * 16);
    float4*   ND   = (float4*)carve((size_t)N * 16);
    float*    DEG  = (float*)carve((size_t)N * 4);
    float*    DINV = (float*)carve((size_t)N * 4);
    float*    WLN  = (float*)carve(2 * 9 * 4);
    float*    WRN  = (float*)carve(2 * 1024 * 4);
    _Float16* B1SW = (_Float16*)carve(4 * 6 * 512 * 2);
    _Float16* B2SW = (_Float16*)carve(3 * 2 * 512 * 2);
    _Float16* BRSW = (_Float16*)carve(2 * 1 * 2 * 512 * 2);
    if ((size_t)(p - (char*)d_ws) > ws_size) return;

    const int TB = 256;

    // ---- one-time prep (recomputed every call; deterministic) ----
    cvt_f32_to_f16<<<cdiv((long)N * IN_CH, TB), TB, 0, stream>>>(x, XH, N * IN_CH);
    specnorm_kernel<<<2, 32, 0, stream>>>(Wl, Wr, WLN, WRN);
    swizzle_b_f16<<<cdiv(4 * 6 * 512, TB), TB, 0, stream>>>(W1, B1SW, IN_CH, 4, 6);
    swizzle_b_f16<<<cdiv(3 * 2 * 512, TB), TB, 0, stream>>>(W2, B2SW, FH, 3, 2);
    swizzle_b_f16<<<cdiv(1 * 2 * 512, TB), TB, 0, stream>>>(WRN, BRSW, HID, 1, 2);
    swizzle_b_f16<<<cdiv(1 * 2 * 512, TB), TB, 0, stream>>>(WRN + 1024, BRSW + 1024, HID, 1, 2);

    // ---- lin1 + ELU: XC = ELU(x @ W1^T + b1)  [N,96] ----
    gemm_wmma_f16<4, 6, 1><<<N / 16, 32, 0, stream>>>(XH, B1SW, b1, XC);

    for (int l = 0; l < 2; ++l) {
        fill_f32<<<cdiv(N, TB), TB, 0, stream>>>(DEG, N, 0.f);
        fill_f32<<<cdiv((long)N * FH, TB), TB, 0, stream>>>(AGG, N * FH, 0.f);

        // factored sheaf/weight learner: per-node dots, then per-edge combine
        node_dots<<<cdiv(N, 4), 128, 0, stream>>>(
            XC, Wsh + (size_t)(l * 3 + 1) * 192, Wwt + (size_t)l * 192, ND, N);
        edge_prep<<<cdiv(NE, TB), TB, 0, stream>>>(ei, ND, EDG, DEG, NE);
        dinv_kernel<<<cdiv(N, TB), TB, 0, stream>>>(DEG, DINV, N);

        // y = Wl_hat * xc * Wr_hat^T : 3x3 mix (VALU) then 32x32 via WMMA
        wl_mix<<<cdiv((long)N * HID, TB), TB, 0, stream>>>(XC, WLN + l * 9, MXH, N);
        gemm_wmma_f16<1, 2, 0><<<(N * FD) / 16, 32, 0, stream>>>(
            MXH, BRSW + l * 1024, nullptr, Y);

        // block-sparse Laplacian apply + residual
        aggregate<<<cdiv((long)NE * 8, TB), TB, 0, stream>>>(ei, EDG, DINV, Y, AGG, NE);
        combine<<<cdiv((long)N * FH, TB), TB, 0, stream>>>(DEG, Y, AGG, eps + l * 3, XC, N);
    }

    // ---- lin2: out = xc @ W2^T + b2  [N,32] ----
    cvt_f32_to_f16<<<cdiv((long)N * FH, TB), TB, 0, stream>>>(XC, XH, N * FH);
    gemm_wmma_f16<3, 2, 2><<<N / 16, 32, 0, stream>>>(XH, B2SW, b2, out);
}
